// NT_Xent_loss_69526930587880
// MI455X (gfx1250) — matrix operand order repaired and checked by reading, not compiled, not run on hardware
//
#include <hip/hip_runtime.h>

// ---------------------------------------------------------------------------
// NT-Xent loss, fused: normalize(f32->f16) -> WMMA GEMM + online exp-row-sum
// -> per-row pos/lse term -> mean reduce.  sim matrix is never materialized.
// ---------------------------------------------------------------------------

typedef _Float16 half16 __attribute__((ext_vector_type(16)));
typedef _Float16 half8  __attribute__((ext_vector_type(8)));
typedef _Float16 half4  __attribute__((ext_vector_type(4)));
typedef float    f32x8  __attribute__((ext_vector_type(8)));

#define NROWS   4096
#define TWO_N   8192
#define DDIM    512
#define INV_T   2.0f        // 1 / TEMP, TEMP = 0.5

#define BM      128
#define BN      128
#define KC      64
#define NKC     (DDIM / KC)
#define KSTR    72          // padded LDS stride (halves) -> conflict-free frags

union alignas(32) Frag {
    half16 v;
    half8  h[2];
};

// ---------------- zero the row-sum accumulator -----------------------------
__global__ void ntx_init(float* __restrict__ S) {
    int i = blockIdx.x * blockDim.x + threadIdx.x;
    if (i < TWO_N) S[i] = 0.0f;
}

// ---------------- L2-normalize rows, emit f16 ------------------------------
// one wave per row; 512 floats -> 16 per lane as float4s
__global__ void ntx_normalize(const float* __restrict__ zi,
                              const float* __restrict__ zj,
                              _Float16* __restrict__ zn) {
    const int row  = blockIdx.x * 8 + (threadIdx.x >> 5);
    const int lane = threadIdx.x & 31;
    const float* src = (row < NROWS) ? (zi + (size_t)row * DDIM)
                                     : (zj + (size_t)(row - NROWS) * DDIM);
    const float4* s4 = reinterpret_cast<const float4*>(src);

    float4 v[4];
    float ss = 0.0f;
#pragma unroll
    for (int i = 0; i < 4; ++i) {
        v[i] = s4[lane + i * 32];
        ss += v[i].x * v[i].x + v[i].y * v[i].y + v[i].z * v[i].z + v[i].w * v[i].w;
    }
#pragma unroll
    for (int m = 1; m < 32; m <<= 1) ss += __shfl_xor(ss, m, 32);

    const float scale = 1.0f / fmaxf(sqrtf(ss), 1e-8f);
    _Float16* dst = zn + (size_t)row * DDIM;
#pragma unroll
    for (int i = 0; i < 4; ++i) {
        half4 h;
        h[0] = (_Float16)(v[i].x * scale);
        h[1] = (_Float16)(v[i].y * scale);
        h[2] = (_Float16)(v[i].z * scale);
        h[3] = (_Float16)(v[i].w * scale);
        *reinterpret_cast<half4*>(&dst[(lane + i * 32) * 4]) = h;
    }
}

// ---------------- fused GEMM + exp row-sum ---------------------------------
// grid (64 n-blocks, 64 m-blocks), 256 threads = 8 waves as 4(M) x 2(N).
// wave (wm,wn) owns rows [gm+32wm, gm+32wm+32) x cols [gn+64wn, gn+64wn+64):
// 2 M-tiles x 4 N-tiles; each B fragment feeds two WMMAs.
__global__ void __launch_bounds__(256)
ntx_gemm_expsum(const _Float16* __restrict__ zn, float* __restrict__ S) {
    __shared__ _Float16 As[BM * KSTR];
    __shared__ _Float16 Bs[BN * KSTR];

    const int tid  = threadIdx.x;
    const int wave = tid >> 5;
    const int wm   = wave & 3;           // 0..3 : M group (32 rows)
    const int wn   = wave >> 2;          // 0..1 : N group (64 cols)
    const int lane = tid & 31;
    const int l16  = lane & 15;
    const int part = lane >> 4;          // 0: lanes 0-15, 1: lanes 16-31
    const int gm   = blockIdx.y * BM;    // row base
    const int gn   = blockIdx.x * BN;    // col base

    f32x8 acc[2][4];
#pragma unroll
    for (int mt = 0; mt < 2; ++mt)
#pragma unroll
        for (int nt = 0; nt < 4; ++nt)
#pragma unroll
            for (int v = 0; v < 8; ++v) acc[mt][nt][v] = 0.0f;

    for (int kc = 0; kc < NKC; ++kc) {
        if (kc) __syncthreads();
        // stage 128x64 f16 tiles of A (rows) and B (cols) into LDS, 16B chunks
#pragma unroll
        for (int i = 0; i < 4; ++i) {
            const int id = tid + i * 256;      // 0..1023
            const int r  = id >> 3;            // 0..127
            const int c8 = (id & 7) * 8;       // half offset in K-chunk
            const uint4* ga = reinterpret_cast<const uint4*>(
                zn + (size_t)(gm + r) * DDIM + kc * KC + c8);
            const uint4* gb = reinterpret_cast<const uint4*>(
                zn + (size_t)(gn + r) * DDIM + kc * KC + c8);
            *reinterpret_cast<uint4*>(&As[r * KSTR + c8]) = *ga;
            *reinterpret_cast<uint4*>(&Bs[r * KSTR + c8]) = *gb;
            if (kc + 1 < NKC) {
                __builtin_prefetch(zn + (size_t)(gm + r) * DDIM + (kc + 1) * KC + c8, 0, 3);
                __builtin_prefetch(zn + (size_t)(gn + r) * DDIM + (kc + 1) * KC + c8, 0, 3);
            }
        }
        __syncthreads();

#pragma unroll
        for (int kb = 0; kb < KC; kb += 32) {
            // A fragments (16x32): lane l16 = row, K split 8+8 at base 8*part
            Frag fa[2];
#pragma unroll
            for (int mt = 0; mt < 2; ++mt) {
                const int arow = wm * 32 + mt * 16 + l16;
                fa[mt].h[0] = *reinterpret_cast<const half8*>(&As[arow * KSTR + kb + part * 8]);
                fa[mt].h[1] = *reinterpret_cast<const half8*>(&As[arow * KSTR + kb + 16 + part * 8]);
            }
#pragma unroll
            for (int nt = 0; nt < 4; ++nt) {
                // B fragment (32x16): lane l16 = col, K split 8+8 at base 8*part
                Frag fb;
                const int bcol = wn * 64 + nt * 16 + l16;
                fb.h[0] = *reinterpret_cast<const half8*>(&Bs[bcol * KSTR + kb + part * 8]);
                fb.h[1] = *reinterpret_cast<const half8*>(&Bs[bcol * KSTR + kb + 16 + part * 8]);
#pragma unroll
                for (int mt = 0; mt < 2; ++mt) {
                    acc[mt][nt] = __builtin_amdgcn_wmma_f32_16x16x32_f16(
                        false, fa[mt].v, false, fb.v, (short)0, acc[mt][nt], false, false);
                }
            }
        }
    }

    // exp + per-row accumulation; C layout: VGPR v -> M = v + 8*part, N = l16
    float rowacc[2][8];
#pragma unroll
    for (int mt = 0; mt < 2; ++mt)
#pragma unroll
        for (int v = 0; v < 8; ++v) rowacc[mt][v] = 0.0f;

#pragma unroll
    for (int mt = 0; mt < 2; ++mt) {
        const int growBase = gm + wm * 32 + mt * 16 + part * 8;
#pragma unroll
        for (int nt = 0; nt < 4; ++nt) {
            const int gcol = gn + wn * 64 + nt * 16 + l16;
#pragma unroll
            for (int v = 0; v < 8; ++v) {
                const float sv = acc[mt][nt][v] * INV_T;   // |sv| <= 2 -> exp safe
                rowacc[mt][v] += (gcol == (growBase + v)) ? 0.0f : __expf(sv);
            }
        }
    }
    // reduce across the 16 lanes of each half (N dimension), then accumulate
#pragma unroll
    for (int mt = 0; mt < 2; ++mt) {
#pragma unroll
        for (int v = 0; v < 8; ++v) {
#pragma unroll
            for (int m = 1; m < 16; m <<= 1)
                rowacc[mt][v] += __shfl_xor(rowacc[mt][v], m, 32);
        }
    }
    if (l16 == 0) {
#pragma unroll
        for (int mt = 0; mt < 2; ++mt) {
            const int growBase = gm + wm * 32 + mt * 16 + part * 8;
#pragma unroll
            for (int v = 0; v < 8; ++v)
                atomicAdd(&S[growBase + v], rowacc[mt][v]);
        }
    }
}

// ---------------- per-row positive + loss term -----------------------------
__global__ void ntx_rowterm(const _Float16* __restrict__ zn,
                            const float* __restrict__ S,
                            float* __restrict__ rowterm) {
    const int row  = blockIdx.x * 8 + (threadIdx.x >> 5);
    const int lane = threadIdx.x & 31;
    const int partner = (row < NROWS) ? row + NROWS : row - NROWS;
    const half4* a = reinterpret_cast<const half4*>(zn + (size_t)row * DDIM);
    const half4* b = reinterpret_cast<const half4*>(zn + (size_t)partner * DDIM);

    float dot = 0.0f;
#pragma unroll
    for (int i = 0; i < 4; ++i) {
        half4 ha = a[lane + i * 32];
        half4 hb = b[lane + i * 32];
#pragma unroll
        for (int c = 0; c < 4; ++c) dot += (float)ha[c] * (float)hb[c];
    }
#pragma unroll
    for (int m = 1; m < 32; m <<= 1) dot += __shfl_xor(dot, m, 32);

    if (lane == 0) {
        const float pos = dot * INV_T;
        // denom = exp(pos) [label logit] + S[row] (off-diag sum, incl. partner)
        rowterm[row] = logf(S[row] + __expf(pos)) - pos;
    }
}

// ---------------- deterministic final mean ---------------------------------
__global__ void ntx_reduce(const float* __restrict__ rowterm, float* __restrict__ out) {
    __shared__ float red[256];
    float s = 0.0f;
    for (int i = threadIdx.x; i < TWO_N; i += 256) s += rowterm[i];
    red[threadIdx.x] = s;
    __syncthreads();
    for (int off = 128; off > 0; off >>= 1) {
        if (threadIdx.x < off) red[threadIdx.x] += red[threadIdx.x + off];
        __syncthreads();
    }
    if (threadIdx.x == 0) out[0] = red[0] * (1.0f / (float)TWO_N);
}

// ---------------------------------------------------------------------------
extern "C" void kernel_launch(void* const* d_in, const int* in_sizes, int n_in,
                              void* d_out, int out_size, void* d_ws, size_t ws_size,
                              hipStream_t stream) {
    (void)in_sizes; (void)n_in; (void)out_size; (void)ws_size;
    const float* zi = (const float*)d_in[0];
    const float* zj = (const float*)d_in[1];
    float* out = (float*)d_out;

    char* ws = (char*)d_ws;
    _Float16* zn   = (_Float16*)ws;                              // 8192*512*2 = 8 MiB
    float*    S    = (float*)(ws + (size_t)TWO_N * DDIM * 2);    // 8192 f32
    float*    term = S + TWO_N;                                  // 8192 f32

    ntx_init     <<<(TWO_N + 255) / 256, 256, 0, stream>>>(S);
    ntx_normalize<<<TWO_N / 8,          256, 0, stream>>>(zi, zj, zn);
    ntx_gemm_expsum<<<dim3(TWO_N / BN, TWO_N / BM), 256, 0, stream>>>(zn, S);
    ntx_rowterm  <<<TWO_N / 8,          256, 0, stream>>>(zn, S, term);
    ntx_reduce   <<<1,                  256, 0, stream>>>(term, out);
}
